// TransformerEncoder_79568564125730
// MI455X (gfx1250) — matrix operand order repaired and checked by reading, not compiled, not run on hardware
//
#include <hip/hip_runtime.h>
#include <math.h>

// ---------------- constants (match reference) ----------------
#define Bb 2
#define Ss 1024
#define Ee 1024
#define Hh 16
#define HD 64
#define Ll 6
#define FFd 4096

typedef __attribute__((ext_vector_type(16))) __bf16 v16bf;
typedef __attribute__((ext_vector_type(8)))  __bf16 v8bf;
typedef __attribute__((ext_vector_type(8)))  float  v8f;

union Frag16 { v16bf v; v8bf h[2]; };

// =====================================================================
// Embedding: x[b,s,:] = tok_emb[tokens[b,s]] + pos_emb[s];  also bf16 copy
// grid = B*S blocks of 256 threads, 4 floats/thread
// =====================================================================
__global__ void embed_kernel(const int* __restrict__ tokens,
                             const float* __restrict__ tok_emb,
                             const float* __restrict__ pos_emb,
                             float* __restrict__ x, __bf16* __restrict__ xb)
{
    int t   = blockIdx.x;          // 0..B*S-1
    int s   = t & (Ss - 1);
    int tok = tokens[t];
    int e   = threadIdx.x * 4;
    const float4 tv = *(const float4*)(tok_emb + (size_t)tok * Ee + e);
    const float4 pv = *(const float4*)(pos_emb + (size_t)s   * Ee + e);
    float4 v = make_float4(tv.x + pv.x, tv.y + pv.y, tv.z + pv.z, tv.w + pv.w);
    size_t base = (size_t)t * Ee + e;
    *(float4*)(x + base) = v;
    xb[base + 0] = (__bf16)v.x; xb[base + 1] = (__bf16)v.y;
    xb[base + 2] = (__bf16)v.z; xb[base + 3] = (__bf16)v.w;
}

// =====================================================================
// Transpose + f32->bf16 convert:  out[c][r] = (bf16) in[r][c]
// in: rows x cols (row-major f32);  out: cols x rows (row-major bf16)
// grid = (cols/32, rows/32), block = (32,8)  — LDS tiled
// =====================================================================
__global__ void transpose_to_bf16(const float* __restrict__ in,
                                  __bf16* __restrict__ out,
                                  int rows, int cols)
{
    __shared__ float tile[32][33];
    int c  = blockIdx.x * 32 + threadIdx.x;
    int r0 = blockIdx.y * 32 + threadIdx.y;
#pragma unroll
    for (int i = 0; i < 32; i += 8)
        tile[threadIdx.y + i][threadIdx.x] = in[(size_t)(r0 + i) * cols + c];
    __syncthreads();
    int oc = blockIdx.y * 32 + threadIdx.x;   // original row
    int or0 = blockIdx.x * 32 + threadIdx.y;  // original col
#pragma unroll
    for (int i = 0; i < 32; i += 8)
        out[(size_t)(or0 + i) * rows + oc] = (__bf16)tile[threadIdx.x][threadIdx.y + i];
}

// =====================================================================
// WMMA bf16 GEMM:  C[z] = A[z] (MxK, row-major, stride lda)
//                        @ Bt[z]^T (Bt is NxK row-major, stride ldb) + bias
// z-batch offsets: off = (z/zdiv)*s1 + (z%zdiv)*s2 for A, Bt, C.
// Block = 128 threads (4 waves); wave tile = 64x64 (4x4 wmma frags).
// grid = (N/64, M/256, Z).  M mult of 256, N mult of 64 per launch; K mult of 32.
// MODE: 0=f32 store (val*scale+bias), 1=bf16 store, 2=bf16 relu store,
//       3=bf16 transposed store (C[col*ldc+row])
// =====================================================================
template <int MODE>
__global__ __launch_bounds__(128)
void gemm_wmma_bf16(const __bf16* __restrict__ A, size_t As1, size_t As2,
                    const __bf16* __restrict__ Bt, size_t Bs1, size_t Bs2,
                    const float* __restrict__ bias,
                    void* __restrict__ Cp, size_t Cs1, size_t Cs2,
                    int zdiv, int K, int lda, int ldb, int ldc, float scale)
{
    const int z  = blockIdx.z;
    const int zq = z / zdiv, zr = z % zdiv;
    A  += (size_t)zq * As1 + (size_t)zr * As2;
    Bt += (size_t)zq * Bs1 + (size_t)zr * Bs2;

    const int lane = threadIdx.x & 31;
    const int wid  = threadIdx.x >> 5;
    const int m0   = blockIdx.y * 256 + wid * 64;
    const int n0   = blockIdx.x * 64;
    const int lm   = lane & 15;      // row/col within fragment
    const int lh   = lane >> 4;      // lane-half

    v8f acc[4][4] = {};

    for (int k0 = 0; k0 < K; k0 += 32) {
        Frag16 af[4], bf[4];
#pragma unroll
        for (int i = 0; i < 4; ++i) {
            // A fragment: lane holds M=m0+i*16+lm; K = lh*8 + e (e<8), +16 for e>=8
            const __bf16* pa = A + (size_t)(m0 + i * 16 + lm) * lda + k0 + lh * 8;
            af[i].h[0] = *(const v8bf*)(pa);
            af[i].h[1] = *(const v8bf*)(pa + 16);
            // B fragment: lane holds N=n0+i*16+lm; K = lh*8 + e (e<8), +16 for e>=8
            const __bf16* pb = Bt + (size_t)(n0 + i * 16 + lm) * ldb + k0 + lh * 8;
            bf[i].h[0] = *(const v8bf*)(pb);
            bf[i].h[1] = *(const v8bf*)(pb + 16);
        }
#pragma unroll
        for (int i = 0; i < 4; ++i)
#pragma unroll
            for (int j = 0; j < 4; ++j)
                acc[i][j] = __builtin_amdgcn_wmma_f32_16x16x32_bf16(
                    false, af[i].v, false, bf[j].v, (short)0, acc[i][j], false, false);
    }

    const size_t coff = (size_t)zq * Cs1 + (size_t)zr * Cs2;
#pragma unroll
    for (int i = 0; i < 4; ++i)
#pragma unroll
        for (int j = 0; j < 4; ++j) {
            const int col = n0 + j * 16 + lm;
            const float bv = bias ? bias[col] : 0.0f;
#pragma unroll
            for (int r = 0; r < 8; ++r) {
                const int row = m0 + i * 16 + lh * 8 + r;
                float v = acc[i][j][r] * scale + bv;
                if (MODE == 0)
                    ((float*)Cp)[coff + (size_t)row * ldc + col] = v;
                else if (MODE == 1)
                    ((__bf16*)Cp)[coff + (size_t)row * ldc + col] = (__bf16)v;
                else if (MODE == 2)
                    ((__bf16*)Cp)[coff + (size_t)row * ldc + col] = (__bf16)fmaxf(v, 0.0f);
                else  // MODE 3: transposed store (for V^T)
                    ((__bf16*)Cp)[coff + (size_t)col * ldc + row] = (__bf16)v;
            }
        }
}

// =====================================================================
// Masked softmax over rows of length S, in place on f32 scores, plus a
// bf16 copy.  grid = B*H*S blocks of 256 threads (4 elems/thread).
// =====================================================================
__global__ void softmax_kernel(float* __restrict__ scores,
                               __bf16* __restrict__ scores_bf,
                               const int* __restrict__ mask)
{
    __shared__ float red[256];
    const int   row  = blockIdx.x;             // b*H*S + h*S + q
    const int   b    = row / (Hh * Ss);
    const int   tid  = threadIdx.x;
    float* p = scores + (size_t)row * Ss;
    float4 v = ((float4*)p)[tid];
    const int4 m = ((const int4*)(mask + (size_t)b * Ss))[tid];
    if (m.x == 0) v.x = -1.0e9f;
    if (m.y == 0) v.y = -1.0e9f;
    if (m.z == 0) v.z = -1.0e9f;
    if (m.w == 0) v.w = -1.0e9f;

    float mx = fmaxf(fmaxf(v.x, v.y), fmaxf(v.z, v.w));
    red[tid] = mx; __syncthreads();
    for (int st = 128; st > 0; st >>= 1) {
        if (tid < st) red[tid] = fmaxf(red[tid], red[tid + st]);
        __syncthreads();
    }
    mx = red[0]; __syncthreads();

    v.x = __expf(v.x - mx); v.y = __expf(v.y - mx);
    v.z = __expf(v.z - mx); v.w = __expf(v.w - mx);
    red[tid] = v.x + v.y + v.z + v.w; __syncthreads();
    for (int st = 128; st > 0; st >>= 1) {
        if (tid < st) red[tid] += red[tid + st];
        __syncthreads();
    }
    const float inv = 1.0f / red[0];
    v.x *= inv; v.y *= inv; v.z *= inv; v.w *= inv;

    ((float4*)p)[tid] = v;
    __bf16* pb = scores_bf + (size_t)row * Ss + tid * 4;
    pb[0] = (__bf16)v.x; pb[1] = (__bf16)v.y; pb[2] = (__bf16)v.z; pb[3] = (__bf16)v.w;
}

// =====================================================================
// Residual add + LayerNorm: out = LN(a+b)*g+be; f32 + bf16 outputs.
// grid = B*S blocks of 256 threads (E=1024, 4 elems/thread)
// =====================================================================
__global__ void add_ln_kernel(const float* __restrict__ a, const float* __restrict__ b,
                              const float* __restrict__ g, const float* __restrict__ be,
                              float* __restrict__ outf, __bf16* __restrict__ outb)
{
    __shared__ float red[256];
    const int row = blockIdx.x;
    const int tid = threadIdx.x;
    const size_t base = (size_t)row * Ee;
    float4 av = ((const float4*)(a + base))[tid];
    float4 bv = ((const float4*)(b + base))[tid];
    float x0 = av.x + bv.x, x1 = av.y + bv.y, x2 = av.z + bv.z, x3 = av.w + bv.w;

    red[tid] = x0 + x1 + x2 + x3; __syncthreads();
    for (int st = 128; st > 0; st >>= 1) {
        if (tid < st) red[tid] += red[tid + st];
        __syncthreads();
    }
    const float mu = red[0] * (1.0f / (float)Ee); __syncthreads();

    float d0 = x0 - mu, d1 = x1 - mu, d2 = x2 - mu, d3 = x3 - mu;
    red[tid] = d0 * d0 + d1 * d1 + d2 * d2 + d3 * d3; __syncthreads();
    for (int st = 128; st > 0; st >>= 1) {
        if (tid < st) red[tid] += red[tid + st];
        __syncthreads();
    }
    const float inv = rsqrtf(red[0] * (1.0f / (float)Ee) + 1e-5f);

    const int e = tid * 4;
    float o0 = d0 * inv * g[e + 0] + be[e + 0];
    float o1 = d1 * inv * g[e + 1] + be[e + 1];
    float o2 = d2 * inv * g[e + 2] + be[e + 2];
    float o3 = d3 * inv * g[e + 3] + be[e + 3];
    *(float4*)(outf + base + e) = make_float4(o0, o1, o2, o3);
    outb[base + e + 0] = (__bf16)o0; outb[base + e + 1] = (__bf16)o1;
    outb[base + e + 2] = (__bf16)o2; outb[base + e + 3] = (__bf16)o3;
}

// =====================================================================
// Host-side orchestration
// =====================================================================
extern "C" void kernel_launch(void* const* d_in, const int* in_sizes, int n_in,
                              void* d_out, int out_size, void* d_ws, size_t ws_size,
                              hipStream_t stream)
{
    (void)in_sizes; (void)n_in; (void)out_size; (void)ws_size;

    const int*   tokens  = (const int*)  d_in[0];
    const int*   amask   = (const int*)  d_in[1];
    const float* tok_emb = (const float*)d_in[2];
    const float* pos_emb = (const float*)d_in[3];
    const float* Wq = (const float*)d_in[4];  const float* bq = (const float*)d_in[5];
    const float* Wk = (const float*)d_in[6];  const float* bk = (const float*)d_in[7];
    const float* Wv = (const float*)d_in[8];  const float* bv = (const float*)d_in[9];
    const float* Wo = (const float*)d_in[10]; const float* bo = (const float*)d_in[11];
    const float* W1 = (const float*)d_in[12]; const float* b1 = (const float*)d_in[13];
    const float* W2 = (const float*)d_in[14]; const float* b2 = (const float*)d_in[15];
    const float* g1 = (const float*)d_in[16]; const float* be1 = (const float*)d_in[17];
    const float* g2 = (const float*)d_in[18]; const float* be2 = (const float*)d_in[19];

    // ---- workspace carve-up ----
    const size_t NT = (size_t)Bb * Ss;           // 2048 tokens
    size_t off = 0;
    auto carve = [&](size_t bytes) {
        void* p = (char*)d_ws + off;
        off += (bytes + 255) & ~(size_t)255;
        return p;
    };
    float*  x    = (float*)  carve(NT * Ee * 4);          // activations
    __bf16* xb   = (__bf16*) carve(NT * Ee * 2);          // bf16 of x
    __bf16* qb   = (__bf16*) carve(NT * Ee * 2);
    __bf16* kb   = (__bf16*) carve(NT * Ee * 2);
    __bf16* vt   = (__bf16*) carve((size_t)Bb * Ee * Ss * 2);  // V^T per (b,h): [d][s]
    __bf16* ob   = (__bf16*) carve(NT * Ee * 2);          // attention output bf16
    float*  tmp  = (float*)  carve(NT * Ee * 4);          // Wo / FF2 f32 output
    float*  x1   = (float*)  carve(NT * Ee * 4);
    __bf16* x1b  = (__bf16*) carve(NT * Ee * 2);
    __bf16* ffb  = (__bf16*) carve(NT * FFd * 2);
    __bf16* atb  = (__bf16*) carve((size_t)Bb * Hh * Ss * Ss * 2);  // softmax bf16
    __bf16* wT   = (__bf16*) carve((size_t)(4 * Ee * Ee + 2 * Ee * FFd) * 2);
    __bf16* wqT = wT;                       // E x E
    __bf16* wkT = wqT + (size_t)Ee * Ee;
    __bf16* wvT = wkT + (size_t)Ee * Ee;
    __bf16* woT = wvT + (size_t)Ee * Ee;
    __bf16* w1T = woT + (size_t)Ee * Ee;    // FF x E
    __bf16* w2T = w1T + (size_t)Ee * FFd;   // E x FF

    float* attn_out = (float*)d_out + NT * Ee;  // (L,B,H,S,S) region

    // ---- embedding ----
    embed_kernel<<<NT, 256, 0, stream>>>(tokens, tok_emb, pos_emb, x, xb);

    const dim3 tb(32, 8);
    const size_t SE = (size_t)Ss * Ee, SS = (size_t)Ss * Ss, SF = (size_t)Ss * FFd;

    for (int l = 0; l < Ll; ++l) {
        const size_t wofs = (size_t)l * Ee * Ee;
        const size_t f1   = (size_t)l * Ee * FFd;
        // weight transpose+convert for this layer
        transpose_to_bf16<<<dim3(32, 32), tb, 0, stream>>>(Wq + wofs, wqT, Ee, Ee);
        transpose_to_bf16<<<dim3(32, 32), tb, 0, stream>>>(Wk + wofs, wkT, Ee, Ee);
        transpose_to_bf16<<<dim3(32, 32), tb, 0, stream>>>(Wv + wofs, wvT, Ee, Ee);
        transpose_to_bf16<<<dim3(32, 32), tb, 0, stream>>>(Wo + wofs, woT, Ee, Ee);
        transpose_to_bf16<<<dim3(128, 32), tb, 0, stream>>>(W1 + f1, w1T, Ee, FFd);
        transpose_to_bf16<<<dim3(32, 128), tb, 0, stream>>>(W2 + f1, w2T, FFd, Ee);

        // Q = x@Wq+bq (bf16), K = x@Wk+bk (bf16): z=B, M=S, N=E, K=E
        gemm_wmma_bf16<1><<<dim3(Ee / 64, Ss / 256, Bb), 128, 0, stream>>>(
            xb, SE, 0, wqT, 0, 0, bq + (size_t)l * Ee, qb, SE, 0,
            1, Ee, Ee, Ee, Ee, 1.0f);
        gemm_wmma_bf16<1><<<dim3(Ee / 64, Ss / 256, Bb), 128, 0, stream>>>(
            xb, SE, 0, wkT, 0, 0, bk + (size_t)l * Ee, kb, SE, 0,
            1, Ee, Ee, Ee, Ee, 1.0f);
        // V^T (transposed store): vt[b*E*S + e*S + s]
        gemm_wmma_bf16<3><<<dim3(Ee / 64, Ss / 256, Bb), 128, 0, stream>>>(
            xb, SE, 0, wvT, 0, 0, bv + (size_t)l * Ee, vt, (size_t)Ee * Ss, 0,
            1, Ee, Ee, Ee, Ss, 1.0f);

        // scores = Q@K^T * 1/sqrt(HD) -> d_out attn slab (f32); z=b*H+h
        float* attnL = attn_out + (size_t)l * Bb * Hh * Ss * Ss;
        gemm_wmma_bf16<0><<<dim3(Ss / 64, Ss / 256, Bb * Hh), 128, 0, stream>>>(
            qb, SE, (size_t)HD, kb, SE, (size_t)HD, nullptr,
            attnL, (size_t)Hh * SS, SS,
            Hh, HD, Ee, Ee, Ss, 0.125f);

        // softmax (masked) in place + bf16 copy
        softmax_kernel<<<Bb * Hh * Ss, 256, 0, stream>>>(attnL, atb, amask);

        // O = attn @ V  -> ob (bf16, [token][E] with head offset); N=HD
        gemm_wmma_bf16<1><<<dim3(HD / 64, Ss / 256, Bb * Hh), 128, 0, stream>>>(
            atb, (size_t)Hh * SS, SS, vt, (size_t)Ee * Ss, (size_t)HD * Ss, nullptr,
            ob, SE, (size_t)HD,
            Hh, Ss, Ss, Ss, Ee, 1.0f);

        // proj: tmp = O@Wo + bo (f32)
        gemm_wmma_bf16<0><<<dim3(Ee / 64, Ss / 256, Bb), 128, 0, stream>>>(
            ob, SE, 0, woT, 0, 0, bo + (size_t)l * Ee, tmp, SE, 0,
            1, Ee, Ee, Ee, Ee, 1.0f);

        // x1 = LN(x + tmp)
        add_ln_kernel<<<NT, 256, 0, stream>>>(x, tmp, g1 + (size_t)l * Ee,
                                              be1 + (size_t)l * Ee, x1, x1b);

        // ff = relu(x1@W1+b1) (bf16)
        gemm_wmma_bf16<2><<<dim3(FFd / 64, Ss / 256, Bb), 128, 0, stream>>>(
            x1b, SE, 0, w1T, 0, 0, b1 + (size_t)l * FFd, ffb, SF, 0,
            1, Ee, Ee, Ee, FFd, 1.0f);

        // tmp = ff@W2 + b2 (f32)
        gemm_wmma_bf16<0><<<dim3(Ee / 64, Ss / 256, Bb), 128, 0, stream>>>(
            ffb, SF, 0, w2T, 0, 0, b2 + (size_t)l * Ee, tmp, SE, 0,
            1, FFd, FFd, FFd, Ee, 1.0f);

        // x = LN(x1 + tmp)  (next layer input, f32 + bf16)
        add_ln_kernel<<<NT, 256, 0, stream>>>(x1, tmp, g2 + (size_t)l * Ee,
                                              be2 + (size_t)l * Ee, x, xb);
    }

    // final hidden states -> front of d_out
    hipMemcpyAsync(d_out, x, NT * Ee * sizeof(float),
                   hipMemcpyDeviceToDevice, stream);
}
